// PointNetSetAbstractionCPP_81183471829339
// MI455X (gfx1250) — matrix operand order repaired and checked by reading, not compiled, not run on hardware
//
#include <hip/hip_runtime.h>

// ---------------- problem constants ----------------
#define NB       16
#define NN       8192
#define NPOINT   1024
#define NSAMPLE  32
#define PTS_D    64
#define IN_CH    67          // 3 (rel xyz) + 64 (point features)
#define COLS     (NB * NSAMPLE * NPOINT)   // 524288, col = (b*K + k)*S + s
#define R2       0.16f
#define EPSB     1e-5f
#define XPAD     104         // 96 (max icPad) + 8 halves LDS row padding

typedef __attribute__((ext_vector_type(16))) _Float16 v16h;
typedef __attribute__((ext_vector_type(8)))  _Float16 v8h;
typedef __attribute__((ext_vector_type(8)))  float    v8f;

// ---------------- furthest point sampling ----------------
// one workgroup per batch, 1024 threads (32 wave32s), 8 points/thread in regs
__global__ __launch_bounds__(1024) void fps_kernel(const float* __restrict__ xyz,
                                                   int* __restrict__ fidx,
                                                   float* __restrict__ newxyz)
{
    const int b   = blockIdx.x;
    const int tid = threadIdx.x;
    const int wid = tid >> 5, lane = tid & 31;
    const float* xb = xyz + (size_t)b * 3 * NN;

    float px[8], py[8], pz[8], dmin[8];
    for (int i = 0; i < 8; ++i) {
        int n = i * 1024 + tid;
        px[i] = xb[n]; py[i] = xb[NN + n]; pz[i] = xb[2 * NN + n];
        dmin[i] = 1e10f;
    }

    __shared__ float cx, cy, cz;
    __shared__ int   sFar;
    __shared__ float wv[32];
    __shared__ int   wi[32];

    int far = 0;
    for (int s = 0; s < NPOINT; ++s) {
        if (tid == 0) {
            float fx = xb[far], fy = xb[NN + far], fz = xb[2 * NN + far];
            cx = fx; cy = fy; cz = fz;
            fidx[b * NPOINT + s] = far;
            float* o = newxyz + ((size_t)b * NPOINT + s) * 3;
            o[0] = fx; o[1] = fy; o[2] = fz;
        }
        __syncthreads();
        float bx = cx, by = cy, bz = cz;
        float bestv = -1.0f; int besti = 0x7fffffff;
        for (int i = 0; i < 8; ++i) {
            float dx = px[i] - bx, dy = py[i] - by, dz = pz[i] - bz;
            float d = dx * dx + dy * dy + dz * dz;
            dmin[i] = fminf(dmin[i], d);
            int n = i * 1024 + tid;
            if (dmin[i] > bestv || (dmin[i] == bestv && n < besti)) { bestv = dmin[i]; besti = n; }
        }
        // wave32 argmax reduction (tie -> lowest index, matches jnp.argmax)
        for (int off = 16; off > 0; off >>= 1) {
            float ov = __shfl_xor(bestv, off, 32);
            int   oi = __shfl_xor(besti, off, 32);
            if (ov > bestv || (ov == bestv && oi < besti)) { bestv = ov; besti = oi; }
        }
        if (lane == 0) { wv[wid] = bestv; wi[wid] = besti; }
        __syncthreads();
        if (wid == 0) {
            float v2 = wv[lane]; int i2 = wi[lane];
            for (int off = 16; off > 0; off >>= 1) {
                float ov = __shfl_xor(v2, off, 32);
                int   oi = __shfl_xor(i2, off, 32);
                if (ov > v2 || (ov == v2 && oi < i2)) { v2 = ov; i2 = oi; }
            }
            if (lane == 0) sFar = i2;
        }
        __syncthreads();
        far = sFar;
    }
}

// ---------------- first-k-in-order ball query ----------------
// one thread per centroid; lanes of a wave share the same n each step (broadcast loads)
__global__ void ball_kernel(const float* __restrict__ xyz,
                            const float* __restrict__ newxyz,
                            int* __restrict__ idx)
{
    int t = blockIdx.x * blockDim.x + threadIdx.x;
    if (t >= NB * NPOINT) return;
    int b = t / NPOINT;
    const float* xb = xyz + (size_t)b * 3 * NN;
    const float* c  = newxyz + (size_t)t * 3;
    float cxx = c[0], cyy = c[1], czz = c[2];
    int* out = idx + (size_t)t * NSAMPLE;
    int cnt = 0, first = 0;
    for (int n = 0; n < NN; ++n) {
        float dx = xb[n] - cxx, dy = xb[NN + n] - cyy, dz = xb[2 * NN + n] - czz;
        if (dx * dx + dy * dy + dz * dz < R2) {
            if (cnt == 0) first = n;
            out[cnt++] = n;
            if (cnt == NSAMPLE) break;
        }
    }
    for (; cnt < NSAMPLE; ++cnt) out[cnt] = first;
}

// ---------------- gather + concat -> f16 feature matrix [IN_CH][COLS] ----------------
__global__ void gather_kernel(const float* __restrict__ xyz,
                              const float* __restrict__ pts,
                              const float* __restrict__ newxyz,
                              const int* __restrict__ idx,
                              _Float16* __restrict__ feat)
{
    int c = blockIdx.x * blockDim.x + threadIdx.x;
    if (c >= COLS) return;
    int s = c & (NPOINT - 1);
    int k = (c >> 10) & (NSAMPLE - 1);
    int b = c >> 15;
    int j = idx[((size_t)(b * NPOINT + s)) * NSAMPLE + k];
    const float* ctr = newxyz + ((size_t)b * NPOINT + s) * 3;
    const float* xb  = xyz + (size_t)b * 3 * NN;
    feat[(size_t)0 * COLS + c] = (_Float16)(xb[j]          - ctr[0]);
    feat[(size_t)1 * COLS + c] = (_Float16)(xb[NN + j]     - ctr[1]);
    feat[(size_t)2 * COLS + c] = (_Float16)(xb[2 * NN + j] - ctr[2]);
    const float* pb = pts + (size_t)b * PTS_D * NN + j;
    for (int d = 0; d < PTS_D; ++d)
        feat[(size_t)(3 + d) * COLS + c] = (_Float16)pb[(size_t)d * NN];
}

// ---------------- WMMA GEMM:  Y[oc][COLS] = W[oc][ic] * X[ic][COLS] (+bias) ----------
// 256 threads (8 waves) per block, 128 columns per block, full OC per block.
// Compile-time tile geometry (ICPAD, OC) so both the K loop and the oc-tile loop
// fully unroll and the per-wave B fragments stay in VGPRs (no scratch spill):
// each wave owns ONE 16-column tile (colT = waveId), loads its B fragments from
// LDS once, and reuses them for every oc tile. Only A fragments stream from LDS.
// Optional fused "BN+ReLU of previous layer" on the X staging path (scale/shift).
// Per-channel sum / sum-of-squares accumulated for the *next* BN (f32, via LDS
// ds_add_f32 partials then one global atomic per channel per block).
template<int ICPAD, int OC, bool FINAL>
__global__ __launch_bounds__(256) void gemm_kernel(
    const _Float16* __restrict__ X, const float* __restrict__ scale,
    const float* __restrict__ shift, const float* __restrict__ W,
    const float* __restrict__ bias, int icTot,
    _Float16* __restrict__ Y, float* __restrict__ outF,
    float* __restrict__ gSum, float* __restrict__ gSsq)
{
    constexpr int NKB  = ICPAD / 32;   // K steps of 32 (2 or 3)
    constexpr int NOCT = OC / 16;      // 16x16 output tiles in oc (4 or 8)

    __shared__ __align__(16) _Float16 Wl[OC * 96];      // [oc][96] row-major
    __shared__ __align__(16) _Float16 Xl[128 * XPAD];   // [col][ICPAD] row-major (+pad)
    __shared__ float sSum[OC], sSsq[OC];

    const int tid = threadIdx.x;
    const int wid = tid >> 5, lane = tid & 31;
    const int hl = lane & 15, hi = lane >> 4;

    if (tid < OC) { sSum[tid] = 0.f; sSsq[tid] = 0.f; }

    // stage weights f32 -> f16 (zero-pad K to ICPAD)
    for (int e = tid; e < OC * ICPAD; e += 256) {
        int o = e / ICPAD, ci = e - o * ICPAD;
        Wl[o * 96 + ci] = (ci < icTot) ? (_Float16)W[o * icTot + ci] : (_Float16)0.f;
    }

    const int col0 = blockIdx.x * 128;
    // stage 128-column activation tile (fused BN+ReLU of previous layer)
    for (int e = tid; e < ICPAD * 128; e += 256) {
        int ci = e >> 7, j = e & 127;
        _Float16 v = (_Float16)0.f;
        if (ci < icTot) {
            float f = (float)X[(size_t)ci * COLS + col0 + j];
            if (scale) f = fmaxf(f * scale[ci] + shift[ci], 0.f);
            v = (_Float16)f;
        }
        Xl[j * XPAD + ci] = v;
    }
    __syncthreads();

    const int colBase = wid * 16;      // fixed 16-column tile per wave
    const int col = col0 + colBase + hl;

    // hoist B (activation) fragments into registers: reused across all oc tiles
    union frag16 { v16h v; v8h h[2]; };
    frag16 Bf[NKB];
#pragma unroll
    for (int kb = 0; kb < NKB; ++kb) {
        // B 32x16: lane holds K = kb*32 + hi*16 + [0..15] of column (colBase+hl)
        const v8h* xp = (const v8h*)&Xl[(colBase + hl) * XPAD + kb * 32 + hi * 16];
        Bf[kb].h[0] = xp[0];
        Bf[kb].h[1] = xp[1];
    }

#pragma unroll
    for (int ocT = 0; ocT < NOCT; ++ocT) {   // wave-uniform: EXEC all-ones at WMMA
        const int ocBase = ocT * 16;
        v8f acc = {};
#pragma unroll
        for (int kb = 0; kb < NKB; ++kb) {
            frag16 A;
            // A (weights) 16x32: lane holds K = kb*32 + hi*16 + [0..15]
            const v8h* wp = (const v8h*)&Wl[(ocBase + hl) * 96 + kb * 32 + hi * 16];
            A.h[0] = wp[0];
            A.h[1] = wp[1];   // +8 halves
            acc = __builtin_amdgcn_wmma_f32_16x16x32_f16(
                      false, A.v, false, Bf[kb].v, (short)0, acc, false, false);
        }
        // epilogue: bias + BN stats + store
#pragma unroll
        for (int r = 0; r < 8; ++r) {
            int o = ocBase + r + hi * 8;      // D layout: VGPR r -> M = r + 8*hi
            float v = acc[r] + bias[o];
            atomicAdd(&sSum[o], v);
            atomicAdd(&sSsq[o], v * v);
            if constexpr (FINAL) {  // final layer: f32 straight into d_out [B,128,K,S]
                int s = col & (NPOINT - 1);
                int k = (col >> 10) & (NSAMPLE - 1);
                int b = col >> 15;
                outF[(((size_t)(b * 128 + o)) * NSAMPLE + k) * NPOINT + s] = v;
            } else {                // intermediate: store raw pre-BN activation as f16
                Y[(size_t)o * COLS + col] = (_Float16)v;
            }
        }
    }
    __syncthreads();
    if (tid < OC) {
        atomicAdd(&gSum[tid], sSum[tid]);
        atomicAdd(&gSsq[tid], sSsq[tid]);
    }
}

// ---------------- per-channel BN coefficients from global sums ----------------
__global__ void stats_kernel(const float* gSum, const float* gSsq,
                             const float* __restrict__ g, const float* __restrict__ bb,
                             float* scale, float* shift, int oc)
{
    int t = threadIdx.x;
    if (t < oc) {
        float inv = 1.0f / (float)COLS;
        float mu  = gSum[t] * inv;
        float var = gSsq[t] * inv - mu * mu;
        float sc  = g[t] * rsqrtf(var + EPSB);
        scale[t] = sc;
        shift[t] = bb[t] - mu * sc;
    }
}

// ---------------- final in-place BN + ReLU on d_out x-region ----------------
__global__ void bnrelu_kernel(float* __restrict__ x,
                              const float* __restrict__ scale,
                              const float* __restrict__ shift)
{
    size_t i = (size_t)blockIdx.x * blockDim.x + threadIdx.x;
    if (i >= (size_t)128 * COLS) return;
    int o = (int)((i >> 15) & 127);   // K*S = 32768 = 2^15
    float v = x[i];
    x[i] = fmaxf(v * scale[o] + shift[o], 0.f);
}

__global__ void zero_kernel(float* p, int n)
{
    int i = blockIdx.x * blockDim.x + threadIdx.x;
    if (i < n) p[i] = 0.f;
}

// ---------------- host-side orchestration ----------------
extern "C" void kernel_launch(void* const* d_in, const int* in_sizes, int n_in,
                              void* d_out, int out_size, void* d_ws, size_t ws_size,
                              hipStream_t stream)
{
    const float* xyz = (const float*)d_in[0];
    const float* pts = (const float*)d_in[1];
    const float* w1  = (const float*)d_in[2];
    const float* w2  = (const float*)d_in[3];
    const float* w3  = (const float*)d_in[4];
    const float* b1  = (const float*)d_in[5];
    const float* b2  = (const float*)d_in[6];
    const float* b3  = (const float*)d_in[7];
    const float* g1  = (const float*)d_in[8];
    const float* g2  = (const float*)d_in[9];
    const float* g3  = (const float*)d_in[10];
    const float* bb1 = (const float*)d_in[11];
    const float* bb2 = (const float*)d_in[12];
    const float* bb3 = (const float*)d_in[13];
    (void)in_sizes; (void)n_in; (void)out_size; (void)ws_size;

    char* ws = (char*)d_ws;
    int*   fidx  = (int*)(ws + 0);                               // 64 KB
    int*   idx   = (int*)(ws + (1u << 16));                      // 2 MB
    float* stats = (float*)(ws + 0x220000);                      // 3 * 512 f32
    _Float16* feat0 = (_Float16*)(ws + 0x400000);                // 67*COLS f16 (~67 MB)
    _Float16* y1 = (_Float16*)(ws + 0x400000 + 72ull * 1024 * 1024);   // 64*COLS f16
    _Float16* y2 = (_Float16*)(ws + 0x400000 + 140ull * 1024 * 1024);  // 64*COLS f16

    float* newxyz = (float*)d_out;                   // [B, S, 3]
    float* xout   = (float*)d_out + NB * NPOINT * 3; // [B, 128, K, S]

    // per-layer stats slots: sum, ssq, scale, shift (128 f32 each)
    float* s1 = stats;        float* s2 = stats + 512;  float* s3 = stats + 1024;

    zero_kernel<<<(1536 + 255) / 256, 256, 0, stream>>>(stats, 1536);
    fps_kernel<<<NB, 1024, 0, stream>>>(xyz, fidx, newxyz);
    ball_kernel<<<(NB * NPOINT + 255) / 256, 256, 0, stream>>>(xyz, newxyz, idx);
    gather_kernel<<<COLS / 256, 256, 0, stream>>>(xyz, pts, newxyz, idx, feat0);

    // layer 1: 67 -> 64 (K padded to 96)
    gemm_kernel<96, 64, false><<<COLS / 128, 256, 0, stream>>>(
        feat0, nullptr, nullptr, w1, b1, IN_CH, y1, nullptr, s1, s1 + 128);
    stats_kernel<<<1, 128, 0, stream>>>(s1, s1 + 128, g1, bb1, s1 + 256, s1 + 384, 64);

    // layer 2: 64 -> 64, fused BN1+ReLU on staging
    gemm_kernel<64, 64, false><<<COLS / 128, 256, 0, stream>>>(
        y1, s1 + 256, s1 + 384, w2, b2, 64, y2, nullptr, s2, s2 + 128);
    stats_kernel<<<1, 128, 0, stream>>>(s2, s2 + 128, g2, bb2, s2 + 256, s2 + 384, 64);

    // layer 3: 64 -> 128, fused BN2+ReLU; writes f32 into d_out layout directly
    gemm_kernel<64, 128, true><<<COLS / 128, 256, 0, stream>>>(
        y2, s2 + 256, s2 + 384, w3, b3, 64, nullptr, xout, s3, s3 + 128);
    stats_kernel<<<1, 128, 0, stream>>>(s3, s3 + 128, g3, bb3, s3 + 256, s3 + 384, 128);

    // final BN3 + ReLU in place on d_out
    bnrelu_kernel<<<(unsigned)((128ull * COLS) / 256), 256, 0, stream>>>(xout, s3 + 256, s3 + 384);
}
